// AttentionBlock3_14345190768933
// MI455X (gfx1250) — matrix operand order repaired and checked by reading, not compiled, not run on hardware
//
#include <hip/hip_runtime.h>

// ---------- CDNA5 WMMA types ----------
typedef __attribute__((ext_vector_type(16))) __bf16         v16bf;
typedef __attribute__((ext_vector_type(8)))  float          v8f;
typedef __attribute__((ext_vector_type(8)))  unsigned short u16x8;
typedef __attribute__((ext_vector_type(16))) unsigned short u16x16;

#define WMMA_BF16(A, B, C) \
  __builtin_amdgcn_wmma_f32_16x16x32_bf16(false, (A), false, (B), (short)0, (C), false, false)

// ---------- problem constants (match reference) ----------
#define BATCH   2
#define NNODES  32768            // N per batch
#define C_IN    512
#define C_EMB   128
#define IMG_W   128
#define HW      16384            // 128*128
#define NT      32               // nodes per workgroup (2 x 16-wide N-subtiles)
#define TILES_PER_B (NNODES / NT)   // 1024
#define XLD     520              // LDS stride (ushort) for X: 1040B=260 words, 260%64==4 -> bank-staggered b128 loads
#define OLD     136              // LDS stride (ushort) for O: 272B=68 words,  68%64==4

// weight scratch layout in d_ws (bf16): 5x[128,512] then Wo[128,128]
#define WSZ     (C_EMB * C_IN)   // 65536
#define WO_OFF  (5 * WSZ)

__device__ __forceinline__ unsigned short f2bf(float f) {
  unsigned int u = __builtin_bit_cast(unsigned int, f);
  unsigned int r = u + 0x7FFFu + ((u >> 16) & 1u);   // round-to-nearest-even
  return (unsigned short)(r >> 16);
}

// A operand: 16x32 bf16, row-major source. lane%16 = M row (handled by caller via wrow),
// lanes>=16 take K offset +8 (VGPR0-3) / +24 (VGPR4-7) per ISA table.
__device__ __forceinline__ v16bf mkA(const unsigned short* __restrict__ wrow, int k0, int aoff) {
  u16x8 lo = *(const u16x8*)(wrow + k0 + aoff);        // K = k0+aoff .. +7   -> VGPR0..3
  u16x8 hi = *(const u16x8*)(wrow + k0 + 16 + aoff);   // K = k0+16+aoff ..   -> VGPR4..7
  u16x16 t = __builtin_shufflevector(lo, hi, 0,1,2,3,4,5,6,7,8,9,10,11,12,13,14,15);
  return __builtin_bit_cast(v16bf, t);
}

// B operand: 32x16 bf16 from LDS, X stored node-major [n][k] (k contiguous).
// lane%16 = column n; lanes>=16 take K offset +8 (VGPR0-3) / +24 (VGPR4-7) per ISA table.
__device__ __forceinline__ v16bf mkB(const unsigned short* x, int ld, int k0, int lane) {
  int n  = lane & 15;
  int kb = k0 + ((lane & 16) ? 8 : 0);
  const unsigned short* p = x + n * ld + kb;
  u16x8 lo = *(const u16x8*)(p);
  u16x8 hi = *(const u16x8*)(p + 16);
  u16x16 t = __builtin_shufflevector(lo, hi, 0,1,2,3,4,5,6,7,8,9,10,11,12,13,14,15);
  return __builtin_bit_cast(v16bf, t);
}

// C/D tile store to LDS result buffer res[n*128 + e] (fp32)
__device__ __forceinline__ void store_res(float* res, int mt, int ntile, int lane, v8f acc) {
  int n    = ntile * 16 + (lane & 15);
  int mrow = (lane & 16) ? 8 : 0;
#pragma unroll
  for (int j = 0; j < 8; ++j)
    res[n * C_EMB + mt * 16 + mrow + j] = acc[j];
}

// ---------- helper kernels ----------
__global__ void zero_f32_kernel(float* __restrict__ p, int n) {
  int i = blockIdx.x * blockDim.x + threadIdx.x;
  if (i < n) p[i] = 0.f;
}

__global__ void f32_to_bf16_kernel(const float* __restrict__ s, unsigned short* __restrict__ d, int n) {
  int i = blockIdx.x * blockDim.x + threadIdx.x;
  if (i < n) d[i] = f2bf(s[i]);
}

// ---------- fused attention kernel ----------
// grid = BATCH * TILES_PER_B blocks of 256 threads (8 wave32)
__global__ __launch_bounds__(256) void attn_fused_kernel(
    const float* __restrict__ vfeat,          // [B,512,128,128]
    const float* __restrict__ rfeat,          // [B,512,128,128]
    const unsigned short* __restrict__ wbf,   // bf16 weights in ws
    const float* __restrict__ Wpos,           // [128,2] fp32
    const int* __restrict__ v2p,              // [B,N,2]
    const int* __restrict__ r2p,              // [B,N,2]
    float* __restrict__ out)                  // [B,128,128,128], pre-zeroed
{
  __shared__ alignas(32) unsigned short sX[NT * XLD];        // gathered columns (bf16)  ~33KB
  __shared__ alignas(32) unsigned short sO[NT * OLD];        // attention output (bf16)  ~8.5KB
  __shared__ float sRes[5 * NT * C_EMB];                     // Q,KR,VR,KV,VV (fp32)     80KB
  __shared__ int   sVlin[NT], sRlin[NT];
  __shared__ float sC0[NT], sC1[NT];

  const int tid  = threadIdx.x;
  const int lane = tid & 31;
  const int wave = tid >> 5;                  // 0..7 == M-tile index
  const int bb   = blockIdx.x / TILES_PER_B;  // batch
  const int tile = blockIdx.x % TILES_PER_B;
  const int nodeBase = tile * NT;

  const int arow = wave * 16 + (lane & 15);   // W row for A operand
  const int aoff = (lane & 16) ? 8 : 0;

  // ---- index / coord setup ----
  if (tid < NT) {
    int gn = nodeBase + tid;
    int i0 = v2p[((size_t)bb * NNODES + gn) * 2 + 0];
    int i1 = v2p[((size_t)bb * NNODES + gn) * 2 + 1];
    sVlin[tid] = i0 * IMG_W + i1;
    sC0[tid] = (float)i0;
    sC1[tid] = (float)i1;
  } else if (tid < 2 * NT) {
    int n = tid - NT, gn = nodeBase + n;
    int i0 = r2p[((size_t)bb * NNODES + gn) * 2 + 0];
    int i1 = r2p[((size_t)bb * NNODES + gn) * 2 + 1];
    sRlin[n] = i0 * IMG_W + i1;
  }
  __syncthreads();

  // ---- gather r columns -> LDS (bf16) ----
  {
    const float* __restrict__ rb = rfeat + (size_t)bb * C_IN * HW;
    for (int i = tid; i < NT * C_IN; i += 256) {
      int n = i >> 9, c = i & (C_IN - 1);
      sX[n * XLD + c] = f2bf(rb[(size_t)c * HW + sRlin[n]]);
    }
  }
  __syncthreads();

  // ---- phase 1: Q, KR, VR  -- each A load feeds 2 N-subtiles, each B load feeds 3 matrices ----
  {
    const unsigned short* w0 = wbf + 0 * WSZ + arow * C_IN;
    const unsigned short* w1 = wbf + 1 * WSZ + arow * C_IN;
    const unsigned short* w2 = wbf + 2 * WSZ + arow * C_IN;
    v8f z = {0.f,0.f,0.f,0.f,0.f,0.f,0.f,0.f};
    v8f acc00 = z, acc01 = z, acc10 = z, acc11 = z, acc20 = z, acc21 = z;
#pragma unroll 2
    for (int kk = 0; kk < C_IN; kk += 32) {
      v16bf b0 = mkB(sX, XLD, kk, lane);
      v16bf b1 = mkB(sX + 16 * XLD, XLD, kk, lane);
      v16bf A0 = mkA(w0, kk, aoff);
      acc00 = WMMA_BF16(A0, b0, acc00);
      acc01 = WMMA_BF16(A0, b1, acc01);
      v16bf A1 = mkA(w1, kk, aoff);
      acc10 = WMMA_BF16(A1, b0, acc10);
      acc11 = WMMA_BF16(A1, b1, acc11);
      v16bf A2 = mkA(w2, kk, aoff);
      acc20 = WMMA_BF16(A2, b0, acc20);
      acc21 = WMMA_BF16(A2, b1, acc21);
    }
    store_res(sRes + 0 * NT * C_EMB, wave, 0, lane, acc00);
    store_res(sRes + 0 * NT * C_EMB, wave, 1, lane, acc01);
    store_res(sRes + 1 * NT * C_EMB, wave, 0, lane, acc10);
    store_res(sRes + 1 * NT * C_EMB, wave, 1, lane, acc11);
    store_res(sRes + 2 * NT * C_EMB, wave, 0, lane, acc20);
    store_res(sRes + 2 * NT * C_EMB, wave, 1, lane, acc21);
  }
  __syncthreads();

  // ---- gather v columns (reuse sX) ----
  {
    const float* __restrict__ vb = vfeat + (size_t)bb * C_IN * HW;
    for (int i = tid; i < NT * C_IN; i += 256) {
      int n = i >> 9, c = i & (C_IN - 1);
      sX[n * XLD + c] = f2bf(vb[(size_t)c * HW + sVlin[n]]);
    }
  }
  __syncthreads();

  // ---- phase 2: KV, VV ----
  {
    const unsigned short* w3 = wbf + 3 * WSZ + arow * C_IN;
    const unsigned short* w4 = wbf + 4 * WSZ + arow * C_IN;
    v8f z = {0.f,0.f,0.f,0.f,0.f,0.f,0.f,0.f};
    v8f acc30 = z, acc31 = z, acc40 = z, acc41 = z;
#pragma unroll 2
    for (int kk = 0; kk < C_IN; kk += 32) {
      v16bf b0 = mkB(sX, XLD, kk, lane);
      v16bf b1 = mkB(sX + 16 * XLD, XLD, kk, lane);
      v16bf A3 = mkA(w3, kk, aoff);
      acc30 = WMMA_BF16(A3, b0, acc30);
      acc31 = WMMA_BF16(A3, b1, acc31);
      v16bf A4 = mkA(w4, kk, aoff);
      acc40 = WMMA_BF16(A4, b0, acc40);
      acc41 = WMMA_BF16(A4, b1, acc41);
    }
    store_res(sRes + 3 * NT * C_EMB, wave, 0, lane, acc30);
    store_res(sRes + 3 * NT * C_EMB, wave, 1, lane, acc31);
    store_res(sRes + 4 * NT * C_EMB, wave, 0, lane, acc40);
    store_res(sRes + 4 * NT * C_EMB, wave, 1, lane, acc41);
  }
  __syncthreads();

  // ---- attention: 64 (node,head) pairs x 4 lanes, t=2 softmax ----
  {
    int pr = tid >> 2;               // 0..63
    int n  = pr >> 1, h = pr & 1, lg = tid & 3;
    const float* q  = sRes + 0 * NT * C_EMB + n * C_EMB + h * 64;
    const float* kr = sRes + 1 * NT * C_EMB + n * C_EMB + h * 64;
    const float* vr = sRes + 2 * NT * C_EMB + n * C_EMB + h * 64;
    const float* kv = sRes + 3 * NT * C_EMB + n * C_EMB + h * 64;
    const float* vv = sRes + 4 * NT * C_EMB + n * C_EMB + h * 64;
    float c0 = sC0[n], c1 = sC1[n];
    float sv = 0.f, sr = 0.f;
#pragma unroll
    for (int i = 0; i < 16; ++i) {
      int d = lg * 16 + i, e = h * 64 + d;
      float kvv = kv[d] + Wpos[2 * e] * c0 + Wpos[2 * e + 1] * c1;   // Wpos @ coords
      float qd  = q[d];
      sv += qd * kvv;
      sr += qd * kr[d];
    }
    sv += __shfl_xor(sv, 2, 4); sv += __shfl_xor(sv, 1, 4);
    sr += __shfl_xor(sr, 2, 4); sr += __shfl_xor(sr, 1, 4);
    sv *= 0.125f; sr *= 0.125f;                 // / sqrt(dh=64)
    float mx = fmaxf(sv, sr);
    float ev = __expf(sv - mx), er = __expf(sr - mx);
    float inv = 1.f / (ev + er);
    float av = ev * inv, ar = er * inv;
#pragma unroll
    for (int i = 0; i < 16; ++i) {
      int d = lg * 16 + i;
      sO[n * OLD + h * 64 + d] = f2bf(av * vv[d] + ar * vr[d]);
    }
  }
  __syncthreads();

  // ---- Wo GEMM (K=128), A shared across both N-subtiles, + atomic scatter-add ----
  {
    const unsigned short* worow = wbf + WO_OFF + arow * C_EMB;
    v8f z = {0.f,0.f,0.f,0.f,0.f,0.f,0.f,0.f};
    v8f acc0 = z, acc1 = z;
#pragma unroll
    for (int k0 = 0; k0 < C_EMB; k0 += 32) {
      v16bf a = mkA(worow, k0, aoff);
      acc0 = WMMA_BF16(a, mkB(sO, OLD, k0, lane), acc0);
      acc1 = WMMA_BF16(a, mkB(sO + 16 * OLD, OLD, k0, lane), acc1);
    }
    int mrow = (lane & 16) ? 8 : 0;
    float* ob = out + (size_t)bb * C_EMB * HW;
    int n0 = lane & 15;
    int lin0 = sRlin[n0];
    int lin1 = sRlin[16 + n0];
#pragma unroll
    for (int j = 0; j < 8; ++j) {
      int och = wave * 16 + mrow + j;
      atomicAdd(ob + (size_t)och * HW + lin0, acc0[j]);
      atomicAdd(ob + (size_t)och * HW + lin1, acc1[j]);
    }
  }
}

// ---------- launch ----------
extern "C" void kernel_launch(void* const* d_in, const int* in_sizes, int n_in,
                              void* d_out, int out_size, void* d_ws, size_t ws_size,
                              hipStream_t stream) {
  (void)in_sizes; (void)n_in; (void)ws_size;
  const float* vfeat = (const float*)d_in[0];
  const float* rfeat = (const float*)d_in[1];
  const float* Wq    = (const float*)d_in[2];
  const float* Wkv   = (const float*)d_in[3];
  const float* Wkr   = (const float*)d_in[4];
  const float* Wvv   = (const float*)d_in[5];
  const float* Wvr   = (const float*)d_in[6];
  const float* Wpos  = (const float*)d_in[7];
  const float* Wo    = (const float*)d_in[8];
  const int*   v2p   = (const int*)d_in[9];
  const int*   r2p   = (const int*)d_in[10];
  float* out = (float*)d_out;
  unsigned short* wbf = (unsigned short*)d_ws;   // 688 KB of bf16 weights

  // weight conversion (slot order: Q, KR, VR, KV, VV, Wo)
  f32_to_bf16_kernel<<<(WSZ + 255) / 256, 256, 0, stream>>>(Wq,  wbf + 0 * WSZ, WSZ);
  f32_to_bf16_kernel<<<(WSZ + 255) / 256, 256, 0, stream>>>(Wkr, wbf + 1 * WSZ, WSZ);
  f32_to_bf16_kernel<<<(WSZ + 255) / 256, 256, 0, stream>>>(Wvr, wbf + 2 * WSZ, WSZ);
  f32_to_bf16_kernel<<<(WSZ + 255) / 256, 256, 0, stream>>>(Wkv, wbf + 3 * WSZ, WSZ);
  f32_to_bf16_kernel<<<(WSZ + 255) / 256, 256, 0, stream>>>(Wvv, wbf + 4 * WSZ, WSZ);
  f32_to_bf16_kernel<<<(C_EMB * C_EMB + 255) / 256, 256, 0, stream>>>(Wo, wbf + WO_OFF, C_EMB * C_EMB);

  // zero output (scatter-add target; harness poisons it)
  zero_f32_kernel<<<(out_size + 255) / 256, 256, 0, stream>>>(out, out_size);

  attn_fused_kernel<<<BATCH * TILES_PER_B, 256, 0, stream>>>(
      vfeat, rfeat, wbf, Wpos, v2p, r2p, out);
}
